// MultiHeadAttention_62096637165690
// MI455X (gfx1250) — matrix operand order repaired and checked by reading, not compiled, not run on hardware
//
#include <hip/hip_runtime.h>

// MHA forward, S=2048 B=2 D=1024 H=16 HD=64, gfx1250 (wave32, WMMA, async-LDS).
//   1) cvt: f32->f16 for q/k/v inputs and w_q/w_k/w_v ; mask -> f32 bias
//   2) proj_gemm<0>: Q,K -> [S][B][D] f16 ; proj_gemm<1>: V -> Vt[B][H][HD][S]
//   3) attn: flash attention; K/V^T tiles + bias staged in LDS via async
//      global->LDS b128 copies (ASYNCcnt double buffering), scores computed
//      transposed (S^T = K Q^T) so softmax rows are per-lane.

#define Sdim 2048
#define Bsz  2
#define Dm   1024
#define Hn   16
#define HDim 64
#define Mrows (Sdim * Bsz)   // 4096

typedef __attribute__((ext_vector_type(16))) _Float16 v16h;
typedef __attribute__((ext_vector_type(8)))  _Float16 v8h;
typedef __attribute__((ext_vector_type(8)))  float    v8f;
typedef __attribute__((ext_vector_type(4)))  float    v4f;

// GCC-style vector type matching the async-LDS builtin's parameter type
typedef int b128i __attribute__((vector_size(16)));
typedef __attribute__((address_space(1))) b128i* glob_b128_ptr;
typedef __attribute__((address_space(3))) b128i* lds_b128_ptr;

#ifndef __has_builtin
#define __has_builtin(x) 0
#endif

static __device__ __forceinline__ v16h cat16(v8h lo, v8h hi) {
  v16h r;
#pragma unroll
  for (int i = 0; i < 8; ++i) { r[i] = lo[i]; r[i + 8] = hi[i]; }
  return r;
}

static __device__ __forceinline__ v8f wmma_f16(v16h a, v16h b, v8f c) {
  return __builtin_amdgcn_wmma_f32_16x16x32_f16(false, a, false, b, (short)0, c,
                                                false, false);
}

// ---- async global -> LDS copy of 16 bytes per lane (GLOBAL_LOAD_ASYNC...) --
static __device__ __forceinline__ void async_copy16(const void* g, void* l) {
#if __has_builtin(__builtin_amdgcn_global_load_async_to_lds_b128)
  __builtin_amdgcn_global_load_async_to_lds_b128(
      (glob_b128_ptr)(size_t)g, (lds_b128_ptr)l, 0, 0);
#else
  unsigned loff = (unsigned)(unsigned long long)
      (__attribute__((address_space(3))) void*)l;
  asm volatile("global_load_async_to_lds_b128 %0, %1, off"
               :: "v"(loff), "v"((unsigned long long)(size_t)g)
               : "memory");
#endif
}

template <int N>
static __device__ __forceinline__ void wait_async() {
#if __has_builtin(__builtin_amdgcn_s_wait_asynccnt)
  __builtin_amdgcn_s_wait_asynccnt(N);
#else
  asm volatile("s_wait_asynccnt %0" :: "i"(N) : "memory");
#endif
}

// ---------------------------------------------------------------------------
// 1a) f32 -> f16 conversion, 8 elements / thread
// ---------------------------------------------------------------------------
__global__ void cvt_f32_to_f16(const float* __restrict__ src,
                               _Float16* __restrict__ dst, int n8) {
  int i = blockIdx.x * blockDim.x + threadIdx.x;
  if (i >= n8) return;
  const v4f* s4 = (const v4f*)src + (size_t)i * 2;
  v4f a = s4[0];
  v4f b = s4[1];
  v8h o;
#pragma unroll
  for (int t = 0; t < 4; ++t) {
    o[t]     = (_Float16)a[t];
    o[t + 4] = (_Float16)b[t];
  }
  *((v8h*)dst + i) = o;
}

// 1b) mask (0/1 int) -> additive f32 bias (-1e30 / 0)
__global__ void mask_to_bias(const int* __restrict__ m,
                             float* __restrict__ b, int n) {
  int i = blockIdx.x * blockDim.x + threadIdx.x;
  if (i < n) b[i] = m[i] ? 0.0f : -1.0e30f;
}

// ---------------------------------------------------------------------------
// 2) Projection GEMM:  Y[m,n] = sum_k X[m,k] * W[n,k] + bias[n]
//    8 waves/block; wave tile 16(M) x 64(N); block tile 128 x 64.
// ---------------------------------------------------------------------------
template <int MODE>
__global__ void proj_gemm(const _Float16* __restrict__ X,
                          const _Float16* __restrict__ W,
                          const float* __restrict__ bias,
                          _Float16* __restrict__ Y) {
  const int l  = threadIdx.x & 31;
  const int w  = threadIdx.x >> 5;
  const int li = l & 15;
  const int hf = l >> 4;
  const int m0 = blockIdx.x * 128 + w * 16;
  const int n0 = blockIdx.y * 64;

  v8f acc[4] = {};
  const _Float16* Xr = X + (size_t)(m0 + li) * Dm;

  for (int kb = 0; kb < Dm; kb += 32) {
    v16h a  = *(const v16h*)(Xr + kb + hf * 16);
#pragma unroll
    for (int t = 0; t < 4; ++t) {
      const _Float16* Wr = W + (size_t)(n0 + t * 16 + li) * Dm + kb + hf * 16;
      v16h b = *(const v16h*)Wr;
      acc[t] = wmma_f16(a, b, acc[t]);
    }
  }

#pragma unroll
  for (int t = 0; t < 4; ++t) {
    const int n    = n0 + t * 16 + li;
    const float bn = bias[n];
#pragma unroll
    for (int r = 0; r < 8; ++r) {
      const int m   = m0 + hf * 8 + r;
      const float y = acc[t][r] + bn;
      if (MODE == 0) {
        Y[(size_t)m * Dm + n] = (_Float16)y;
      } else {
        const int s  = m >> 1;
        const int bb = m & 1;
        const int hh = n >> 6;
        const int dd = n & 63;
        Y[((size_t)((bb * Hn + hh) * HDim + dd)) * Sdim + s] = (_Float16)y;
      }
    }
  }
}

// ---------------------------------------------------------------------------
// 3) Flash attention with async-LDS double-buffered K/V^T tiles + LDS bias.
//    Block 256 thr = 8 waves, one (b,h); wave owns a 16-row i-tile.
// ---------------------------------------------------------------------------
__global__ __launch_bounds__(256) void attn_kernel(
    const _Float16* __restrict__ Q,     // [S][B][D] f16
    const _Float16* __restrict__ K,     // [S][B][D] f16
    const _Float16* __restrict__ Vt,    // [B][H][HD][S] f16
    const float* __restrict__ abias,    // [S] additive mask bias
    float* __restrict__ out) {          // [S][B][D] f32
  __shared__ __align__(16) _Float16 sK[2][32 * 64];  // j-rows x d
  __shared__ __align__(16) _Float16 sV[2][64 * 32];  // d-rows x j
  __shared__ __align__(16) float    sBias[Sdim];     // whole mask bias row

  const int tid = threadIdx.x;
  const int l   = tid & 31;
  const int w   = tid >> 5;
  const int li  = l & 15;
  const int hf  = l >> 4;
  const int i0  = blockIdx.x * 128 + w * 16;
  const int bb  = blockIdx.y >> 4;
  const int hh  = blockIdx.y & 15;

  // Q^T B-fragments, pre-scaled by 1/sqrt(HD) = 0.125 (exact in f16)
  const _Float16* Qrow = Q + ((size_t)(i0 + li) * Bsz + bb) * Dm + hh * HDim;
  v16h qb0 = cat16(*(const v8h*)(Qrow + hf * 8), *(const v8h*)(Qrow + hf * 8 + 16));
  v16h qb1 = cat16(*(const v8h*)(Qrow + 32 + hf * 8), *(const v8h*)(Qrow + 32 + hf * 8 + 16));
#pragma unroll
  for (int r = 0; r < 16; ++r) {
    qb0[r] = qb0[r] * (_Float16)0.125f;
    qb1[r] = qb1[r] * (_Float16)0.125f;
  }

  const _Float16* Kbase = K + (size_t)bb * Dm + hh * HDim;
  const _Float16* Vbase = Vt + (size_t)((bb * Hn + hh) * HDim) * Sdim;

  // staging assignments: 256 x 16B chunks per tile, 2 chunks for bias
  const int kjr = tid >> 3, kc = (tid & 7) * 8;
  const int vdr = tid >> 2, vc = (tid & 3) * 8;

  // bias: 2048 f32 = 512 chunks -> 2 per thread
  async_copy16(abias + tid * 8,     &sBias[tid * 8]);
  async_copy16(abias + tid * 8 + 4, &sBias[tid * 8 + 4]);
  // prologue tiles for jt=0
  async_copy16(Kbase + (size_t)kjr * (Bsz * Dm) + kc, &sK[0][kjr * 64 + kc]);
  async_copy16(Vbase + (size_t)vdr * Sdim + vc,       &sV[0][vdr * 32 + vc]);

  float mrow = -3.0e38f;
  float lrow = 0.0f;
  v8f ot[4] = {};

  for (int jt = 0; jt < Sdim; jt += 32) {
    const int buf = (jt >> 5) & 1;
    if (jt + 32 < Sdim) {  // prefetch next tiles into other buffer
      const int nb = buf ^ 1;
      async_copy16(Kbase + (size_t)(jt + 32 + kjr) * (Bsz * Dm) + kc,
                   &sK[nb][kjr * 64 + kc]);
      async_copy16(Vbase + (size_t)vdr * Sdim + (jt + 32) + vc,
                   &sV[nb][vdr * 32 + vc]);
      wait_async<2>();     // everything except the 2 just-issued is complete
    } else {
      wait_async<0>();
    }
    __syncthreads();

    // --- S^T = K * Q^T from LDS ---
    const _Float16* K0 = &sK[buf][li * 64];
    const _Float16* K1 = &sK[buf][(16 + li) * 64];
    v16h ka00 = cat16(*(const v8h*)(K0 + 0  + hf * 8),
                      *(const v8h*)(K0 + 0  + hf * 8 + 16));
    v16h ka01 = cat16(*(const v8h*)(K0 + 32 + hf * 8),
                      *(const v8h*)(K0 + 32 + hf * 8 + 16));
    v16h ka10 = cat16(*(const v8h*)(K1 + 0  + hf * 8),
                      *(const v8h*)(K1 + 0  + hf * 8 + 16));
    v16h ka11 = cat16(*(const v8h*)(K1 + 32 + hf * 8),
                      *(const v8h*)(K1 + 32 + hf * 8 + 16));

    v8f c0 = {}; c0 = wmma_f16(ka00, qb0, c0); c0 = wmma_f16(ka01, qb1, c0);
    v8f c1 = {}; c1 = wmma_f16(ka10, qb0, c1); c1 = wmma_f16(ka11, qb1, c1);

    // --- additive mask bias from LDS; frag VGPR r -> j = jt + hf*8 + r ---
    const float* bp0 = &sBias[jt + hf * 8];
    const float* bp1 = &sBias[jt + 16 + hf * 8];
    v4f b00 = *(const v4f*)(bp0);
    v4f b01 = *(const v4f*)(bp0 + 4);
    v4f b10 = *(const v4f*)(bp1);
    v4f b11 = *(const v4f*)(bp1 + 4);

    float s0[8], s1[8];
#pragma unroll
    for (int r = 0; r < 8; ++r) {
      s0[r] = c0[r] + ((r < 4) ? b00[r] : b01[r - 4]);
      s1[r] = c1[r] + ((r < 4) ? b10[r] : b11[r - 4]);
    }

    // --- online softmax; row i per-lane + one half-swap shuffle ---
    float pm = -3.0e38f;
#pragma unroll
    for (int r = 0; r < 8; ++r) pm = fmaxf(pm, fmaxf(s0[r], s1[r]));
    pm = fmaxf(pm, __shfl_xor(pm, 16, 32));
    const float nm    = fmaxf(mrow, pm);
    const float alpha = __expf(mrow - nm);

    float p0[8], p1[8];
    float rs = 0.0f;
#pragma unroll
    for (int r = 0; r < 8; ++r) {
      p0[r] = __expf(s0[r] - nm);   // masked: -1e30 - nm -> exp underflows to 0
      p1[r] = __expf(s1[r] - nm);
      rs += p0[r] + p1[r];
    }
    rs += __shfl_xor(rs, 16, 32);
    lrow = lrow * alpha + rs;
    mrow = nm;

#pragma unroll
    for (int t = 0; t < 4; ++t)
#pragma unroll
      for (int r = 0; r < 8; ++r) ot[t][r] *= alpha;

    // --- build P^T B-fragment (lane = column i, K = j packed) ---
    v16h pb;
#pragma unroll
    for (int r = 0; r < 8; ++r) {
      const float x0 = __shfl_xor(p0[r], 16, 32);
      const float x1 = __shfl_xor(p1[r], 16, 32);
      pb[r]     = (_Float16)(hf ? x1 : p0[r]);
      pb[r + 8] = (_Float16)(hf ? p1[r] : x0);
    }

    // --- O^T += V^T * P^T from LDS ---
#pragma unroll
    for (int t = 0; t < 4; ++t) {
      const _Float16* Vr = &sV[buf][(t * 16 + li) * 32];
      v16h va = *(const v16h*)(Vr + hf * 16);
      ot[t] = wmma_f16(va, pb, ot[t]);
    }
    __syncthreads();  // protect buffer being refilled next iteration
  }

  const float inv = 1.0f / lrow;
  float* orow = out + ((size_t)(i0 + li) * Bsz + bb) * Dm + hh * HDim;
#pragma unroll
  for (int t = 0; t < 4; ++t) {
    v4f o0, o1;
#pragma unroll
    for (int r = 0; r < 4; ++r) {
      o0[r] = ot[t][r] * inv;
      o1[r] = ot[t][r + 4] * inv;
    }
    *(v4f*)(orow + t * 16 + hf * 8)     = o0;
    *(v4f*)(orow + t * 16 + hf * 8 + 4) = o1;
  }
}

// ---------------------------------------------------------------------------
extern "C" void kernel_launch(void* const* d_in, const int* in_sizes, int n_in,
                              void* d_out, int out_size, void* d_ws,
                              size_t ws_size, hipStream_t stream) {
  (void)in_sizes; (void)n_in; (void)out_size; (void)ws_size;

  const float* q_in = (const float*)d_in[0];
  const float* k_in = (const float*)d_in[1];
  const float* v_in = (const float*)d_in[2];
  const int*   mask = (const int*)d_in[3];
  const float* w_q  = (const float*)d_in[4];
  const float* b_q  = (const float*)d_in[5];
  const float* w_k  = (const float*)d_in[6];
  const float* b_k  = (const float*)d_in[7];
  const float* w_v  = (const float*)d_in[8];
  const float* b_v  = (const float*)d_in[9];
  float* out = (float*)d_out;

  _Float16* ws = (_Float16*)d_ws;
  const size_t NX = (size_t)Mrows * Dm;
  const size_t NW = (size_t)Dm * Dm;
  _Float16* qx  = ws;
  _Float16* kx  = qx + NX;
  _Float16* vx  = kx + NX;
  _Float16* wqh = vx + NX;
  _Float16* wkh = wqh + NW;
  _Float16* wvh = wkh + NW;
  _Float16* Qh  = wvh + NW;
  _Float16* Kh  = Qh + NX;
  _Float16* Vt  = Kh + NX;
  float*    abias = (float*)(Vt + NX);   // [S] f32 additive mask bias

  {
    const int n8 = (int)(NX / 8);
    const int w8 = (int)(NW / 8);
    dim3 b(256);
    dim3 g((n8 + 255) / 256);
    dim3 gw((w8 + 255) / 256);
    cvt_f32_to_f16<<<g, b, 0, stream>>>(q_in, qx, n8);
    cvt_f32_to_f16<<<g, b, 0, stream>>>(k_in, kx, n8);
    cvt_f32_to_f16<<<g, b, 0, stream>>>(v_in, vx, n8);
    cvt_f32_to_f16<<<gw, b, 0, stream>>>(w_q, wqh, w8);
    cvt_f32_to_f16<<<gw, b, 0, stream>>>(w_k, wkh, w8);
    cvt_f32_to_f16<<<gw, b, 0, stream>>>(w_v, wvh, w8);
    mask_to_bias<<<dim3(Sdim / 256), b, 0, stream>>>(mask, abias, Sdim);
  }

  dim3 pb(256), pg(Mrows / 128, Dm / 64);
  proj_gemm<0><<<pg, pb, 0, stream>>>(qx, wqh, b_q, Qh);
  proj_gemm<0><<<pg, pb, 0, stream>>>(kx, wkh, b_k, Kh);
  proj_gemm<1><<<pg, pb, 0, stream>>>(vx, wvh, b_v, Vt);

  dim3 ab(256), ag(Sdim / 128, Bsz * Hn);
  attn_kernel<<<ag, ab, 0, stream>>>(Qh, Kh, Vt, abias, out);
}
